// Attention_83451214561634
// MI455X (gfx1250) — matrix operand order repaired and checked by reading, not compiled, not run on hardware
//
#include <hip/hip_runtime.h>
#include <stdint.h>

typedef __attribute__((ext_vector_type(16))) __bf16 v16bf;
typedef __attribute__((ext_vector_type(8)))  __bf16 v8bf;
typedef __attribute__((ext_vector_type(8)))  float  v8f;

// Problem constants (match reference setup_inputs)
constexpr int T_TOK = 2048;
constexpr int HD    = 4096;
constexpr int NB    = 256;
constexpr int BSZ   = 16;     // cache block size
constexpr int BATCH = 2;
constexpr int BLKS  = 128;    // blocks per sequence
constexpr int H     = 32;
constexpr int D     = 128;    // head dim
constexpr int S     = T_TOK / BATCH;   // 1024 new tokens per seq
constexpr int P     = BLKS * BSZ;      // 2048 kv positions per seq
constexpr int OFF   = P - S;           // 1024 existing context

// LDS layout (byte strides, padded against bank conflicts, 16B aligned)
constexpr int QROWB = 272;             // 128 bf16 (256B) + 16 pad
constexpr int KROWB = 272;             // 128 bf16 + 16 pad
constexpr int VROWB = 80;              // 32 bf16 (64B) + 16 pad
constexpr int PROWB = 80;              // 32 bf16 + 16 pad
constexpr int LDS_Q = 128 * QROWB;     // 34816
constexpr int LDS_K = 32  * KROWB;     // 8704
constexpr int LDS_V = D   * VROWB;     // 10240
constexpr int LDS_P = 8 * 16 * PROWB;  // 10240  (per-wave P scratch)

static __device__ __forceinline__ unsigned pack_bf16(float a, float b) {
    unsigned short lo = __builtin_bit_cast(unsigned short, (__bf16)a);
    unsigned short hi = __builtin_bit_cast(unsigned short, (__bf16)b);
    return (unsigned)lo | ((unsigned)hi << 16);
}

static __device__ __forceinline__ v16bf cat16(v8bf lo, v8bf hi) {
    return __builtin_shufflevector(lo, hi, 0,1,2,3,4,5,6,7,8,9,10,11,12,13,14,15);
}

__global__ __launch_bounds__(256) void paged_attn_prefill_kernel(
    const float* __restrict__ q,
    const float* __restrict__ knew,
    const float* __restrict__ vnew,
    const float* __restrict__ kvc,
    const int*   __restrict__ btab,
    float*       __restrict__ out)
{
    __shared__ __align__(16) unsigned char smem[LDS_Q + LDS_K + LDS_V + LDS_P];
    unsigned char* qs = smem;
    unsigned char* ks = smem + LDS_Q;
    unsigned char* vt = smem + LDS_Q + LDS_K;
    unsigned char* ps = smem + LDS_Q + LDS_K + LDS_V;

    const int tid  = threadIdx.x;
    const int wave = tid >> 5;
    const int lane = tid & 31;
    const int hi   = lane >> 4;   // K-half selector for WMMA operand layouts
    const int ln   = lane & 15;   // row (A) / column (B) within tile

    const int wg      = blockIdx.x;
    const int stripe  = wg & 7;         // which 128-row query stripe
    const int bh      = wg >> 3;
    const int b       = bh >> 5;
    const int h       = bh & 31;
    const int wgRow0  = stripe * 128;
    const int s0      = wgRow0 + wave * 16;       // this wave's query-tile start
    const float scale = 0.08838834764831845f;     // 1/sqrt(128), folded into Q

    const float* kcache = kvc;
    const float* vcache = kvc + (size_t)NB * BSZ * HD;

    // ---- stage this WG's 128 query rows as bf16 into LDS, pre-scaled ----
    {
        const int qr   = tid >> 1;        // 0..127
        const int half = tid & 1;         // 64-dim half
        const float4* src = (const float4*)(q + (size_t)(b * S + wgRow0 + qr) * HD
                                              + h * D + half * 64);
        unsigned* dst = (unsigned*)(qs + qr * QROWB + half * 128);
        #pragma unroll
        for (int i = 0; i < 16; ++i) {
            float4 f = src[i];
            dst[2 * i + 0] = pack_bf16(f.x * scale, f.y * scale);
            dst[2 * i + 1] = pack_bf16(f.z * scale, f.w * scale);
        }
    }
    __syncthreads();

    // ---- load Q fragments (A layout: lane=row ln, elems 0..7 -> K=hi*8+j,
    //      elems 8..15 -> K=16+hi*8+j, per 32-K chunk) into registers ----
    v16bf qfrag[4];
    {
        const unsigned char* qrow = qs + (wave * 16 + ln) * QROWB;
        #pragma unroll
        for (int kc = 0; kc < 4; ++kc) {
            v8bf lo = *(const v8bf*)(qrow + kc * 64 + hi * 16);
            v8bf hh = *(const v8bf*)(qrow + kc * 64 + 32 + hi * 16);
            qfrag[kc] = cat16(lo, hh);
        }
    }

    // ---- accumulators: O (16x128 f32, 8 C-tiles), row stats ----
    v8f   O[8];
    float m_r[8], l_r[8];
    #pragma unroll
    for (int j = 0; j < 8; ++j) { v8f z = {}; O[j] = z; }
    #pragma unroll
    for (int r = 0; r < 8; ++r) { m_r[r] = -__builtin_inff(); l_r[r] = 0.0f; }

    const int nchunks = (OFF + wgRow0 + 128) >> 5;   // 32-key chunks, WG-uniform
    const int kk = tid >> 3;          // staging: key within chunk (0..31)
    const int dp = tid & 7;           // staging: 16-dim group

    for (int c = 0; c < nchunks; ++c) {
        const int p0 = c << 5;
        __syncthreads();              // previous chunk's LDS reads complete

        // ---- cooperative gather of K/V chunk (paged cache + new-token overlay) ----
        {
            const int p = p0 + kk;
            const float* ksrc;
            const float* vsrc;
            if (p >= OFF) {
                size_t rowi = (size_t)(b * S + (p - OFF)) * HD + h * D;
                ksrc = knew + rowi;
                vsrc = vnew + rowi;
            } else {
                int blk = btab[b * BLKS + (p >> 4)];
                size_t rowi = ((size_t)blk * BSZ + (p & 15)) * (size_t)HD + h * D;
                ksrc = kcache + rowi;
                vsrc = vcache + rowi;
            }
            // prefetch next chunk's rows (CDNA5 global_prefetch path)
            {
                const int pn = p + 32;
                if (pn < P) {
                    const float* pf;
                    if (pn >= OFF) {
                        pf = knew + (size_t)(b * S + (pn - OFF)) * HD + h * D;
                    } else {
                        int blk = btab[b * BLKS + (pn >> 4)];
                        pf = kcache + ((size_t)blk * BSZ + (pn & 15)) * (size_t)HD + h * D;
                    }
                    __builtin_prefetch(pf + dp * 16, 0, 3);
                }
            }
            // K: row-major bf16 rows [key][dim]
            const float4* k4 = (const float4*)(ksrc + dp * 16);
            unsigned* kdst = (unsigned*)(ks + kk * KROWB + dp * 32);
            #pragma unroll
            for (int i = 0; i < 4; ++i) {
                float4 f = k4[i];
                kdst[2 * i + 0] = pack_bf16(f.x, f.y);
                kdst[2 * i + 1] = pack_bf16(f.z, f.w);
            }
            // V: transposed bf16 [dim][key]
            const float4* v4 = (const float4*)(vsrc + dp * 16);
            __bf16* vb = (__bf16*)vt;
            #pragma unroll
            for (int i = 0; i < 4; ++i) {
                float4 f = v4[i];
                int d0 = dp * 16 + i * 4;
                vb[(d0 + 0) * (VROWB / 2) + kk] = (__bf16)f.x;
                vb[(d0 + 1) * (VROWB / 2) + kk] = (__bf16)f.y;
                vb[(d0 + 2) * (VROWB / 2) + kk] = (__bf16)f.z;
                vb[(d0 + 3) * (VROWB / 2) + kk] = (__bf16)f.w;
            }
        }
        __syncthreads();

        // waves past their causal frontier just participate in staging/barriers
        if (p0 >= OFF + s0 + 16) continue;

        // ---- QK^T: two 16x16 f32 score tiles over K=128 (4 x k32 WMMAs each) ----
        v8f sc0 = {}, sc1 = {};
        #pragma unroll
        for (int kc = 0; kc < 4; ++kc) {
            // B layout: lane holds column n (=key), elems 0..7 -> K(dim)=hi*8+j, elems 8..15 -> K=16+hi*8+j
            const unsigned char* kr0 = ks + ln * KROWB + kc * 64 + hi * 16;
            v16bf b0 = cat16(*(const v8bf*)(kr0), *(const v8bf*)(kr0 + 32));
            sc0 = __builtin_amdgcn_wmma_f32_16x16x32_bf16(false, qfrag[kc], false, b0,
                                                          (short)0, sc0, false, false);
            const unsigned char* kr1 = ks + (ln + 16) * KROWB + kc * 64 + hi * 16;
            v16bf b1 = cat16(*(const v8bf*)(kr1), *(const v8bf*)(kr1 + 32));
            sc1 = __builtin_amdgcn_wmma_f32_16x16x32_bf16(false, qfrag[kc], false, b1,
                                                          (short)0, sc1, false, false);
        }

        // ---- causal mask, only in the diagonal region (wave-uniform branch) ----
        // C layout: VGPR r, lane -> M = r + 8*hi, N = ln (+16 for tile 1).
        // Any masking possible only when limit < 31.
        const int limit = OFF + s0 - p0;   // key col n visible iff n <= limit + m
        if (limit < 31) {
            #pragma unroll
            for (int r = 0; r < 8; ++r) {
                const int mrow = r + 8 * hi;
                if (ln      > limit + mrow) sc0[r] = -__builtin_inff();
                if (ln + 16 > limit + mrow) sc1[r] = -__builtin_inff();
            }
        }

        // ---- online softmax; emit P tile (bf16) into per-wave LDS scratch ----
        __bf16* pw = (__bf16*)(ps + wave * 16 * PROWB);
        float corr[8];
        #pragma unroll
        for (int r = 0; r < 8; ++r) {
            float mx = fmaxf(sc0[r], sc1[r]);
            mx = fmaxf(mx, __shfl_xor(mx, 1));
            mx = fmaxf(mx, __shfl_xor(mx, 2));
            mx = fmaxf(mx, __shfl_xor(mx, 4));
            mx = fmaxf(mx, __shfl_xor(mx, 8));
            const float mnew = fmaxf(m_r[r], mx);
            const float cr   = __expf(m_r[r] - mnew);
            const float pv0  = __expf(sc0[r] - mnew);
            const float pv1  = __expf(sc1[r] - mnew);
            float rs = pv0 + pv1;
            rs += __shfl_xor(rs, 1);
            rs += __shfl_xor(rs, 2);
            rs += __shfl_xor(rs, 4);
            rs += __shfl_xor(rs, 8);
            l_r[r] = l_r[r] * cr + rs;
            m_r[r] = mnew;
            corr[r] = cr;
            const int mrow = r + 8 * hi;
            pw[mrow * (PROWB / 2) + ln]      = (__bf16)pv0;
            pw[mrow * (PROWB / 2) + ln + 16] = (__bf16)pv1;
        }

        // rescale running output
        #pragma unroll
        for (int j = 0; j < 8; ++j)
            #pragma unroll
            for (int r = 0; r < 8; ++r)
                O[j][r] *= corr[r];

        // ---- reload P in A layout (same-wave LDS, in-order) ----
        const unsigned char* prow = (const unsigned char*)pw + ln * PROWB;
        v16bf pf = cat16(*(const v8bf*)(prow + hi * 16),
                         *(const v8bf*)(prow + 32 + hi * 16));

        // ---- PV: one k32 WMMA per 16-dim output tile ----
        #pragma unroll
        for (int j = 0; j < 8; ++j) {
            const unsigned char* vrow = vt + (16 * j + ln) * VROWB + hi * 16;
            v16bf vf = cat16(*(const v8bf*)(vrow), *(const v8bf*)(vrow + 32));
            O[j] = __builtin_amdgcn_wmma_f32_16x16x32_bf16(false, pf, false, vf,
                                                           (short)0, O[j], false, false);
        }
    }

    // ---- finalize: O /= l, write f32 output ----
    #pragma unroll
    for (int r = 0; r < 8; ++r) {
        const int mrow = r + 8 * hi;
        const float inv = 1.0f / l_r[r];
        const size_t base = (size_t)(b * S + s0 + mrow) * HD + h * D + ln;
        #pragma unroll
        for (int j = 0; j < 8; ++j)
            out[base + 16 * j] = O[j][r] * inv;
    }
}

extern "C" void kernel_launch(void* const* d_in, const int* in_sizes, int n_in,
                              void* d_out, int out_size, void* d_ws, size_t ws_size,
                              hipStream_t stream) {
    const float* q    = (const float*)d_in[0];
    const float* knew = (const float*)d_in[1];
    const float* vnew = (const float*)d_in[2];
    const float* kvc  = (const float*)d_in[3];
    const int*   bt   = (const int*)d_in[4];
    float* out = (float*)d_out;

    dim3 grid(BATCH * H * (S / 128));   // 512 workgroups
    dim3 block(256);                    // 8 waves of 32
    paged_attn_prefill_kernel<<<grid, block, 0, stream>>>(q, knew, vnew, kvc, bt, out);
}
